// GRAPEBase_3049426780537
// MI455X (gfx1250) — compile-verified, matrix operand order checked by reading, not run on hardware
//
#include <hip/hip_runtime.h>
#include <hip/hip_bf16.h>
#include <stdint.h>

typedef __bf16 bf16_t;
typedef __attribute__((ext_vector_type(16))) __bf16 v16bf;
typedef __attribute__((ext_vector_type(8)))  float  v8f;
typedef __attribute__((ext_vector_type(4)))  unsigned u32x4;
typedef __attribute__((ext_vector_type(8)))  int      i32x8;
typedef __attribute__((ext_vector_type(4)))  int      i32x4;

#define HID 128

// ---------------------------------------------------------------------------
// CDNA5 primitives
// ---------------------------------------------------------------------------
static __device__ __forceinline__ v8f wmma_bf16(v16bf a, v16bf b, v8f c) {
  return __builtin_amdgcn_wmma_f32_16x16x32_bf16(false, a, false, b, (short)0, c,
                                                 false, false);
}

static __device__ __forceinline__ void wait_async0() {
#if __has_builtin(__builtin_amdgcn_s_wait_asynccnt)
  __builtin_amdgcn_s_wait_asynccnt(0);
#else
  asm volatile("s_wait_asynccnt 0" ::: "memory");
#endif
}

static __device__ __forceinline__ void wait_tensor0() {
#if __has_builtin(__builtin_amdgcn_s_wait_tensorcnt)
  __builtin_amdgcn_s_wait_tensorcnt(0);
#else
  asm volatile("s_wait_tensorcnt 0" ::: "memory");
#endif
}

// per-lane async copy of 16B global -> LDS (ASYNCcnt-tracked)
static __device__ __forceinline__ void async_b128(void* lds_dst, const void* gsrc) {
  unsigned l = (unsigned)(uintptr_t)lds_dst;
  asm volatile("global_load_async_to_lds_b128 %0, %1, off"
               :: "v"(l), "v"(gsrc) : "memory");
}

// TDM: DMA `elems` bf16 (elems < 65536) from global into LDS as a 1-D tile.
// Issued by one wave; waits TENSORcnt==0 before returning.
static __device__ __forceinline__ void tdm_load_weights(const bf16_t* gsrc,
                                                        void* lds_dst,
                                                        unsigned elems) {
#if __has_builtin(__builtin_amdgcn_tensor_load_to_lds)
  unsigned long long ga = (unsigned long long)(uintptr_t)gsrc;
  // D# group0: count=1 | lds_addr | global_addr[56:0] | type=2
  u32x4 g0 = {1u, (unsigned)(uintptr_t)lds_dst, (unsigned)ga,
              (unsigned)((ga >> 32) & 0x01FFFFFFu) | 0x80000000u};
  // D# group1: data_size=1(2B); tensor_dim0=elems; tensor_dim1=1;
  //            tile_dim0=elems; tensor_dim0_stride=elems
  i32x8 g1 = {(int)(1u << 16),
              (int)((elems & 0xFFFFu) << 16),
              (int)(((elems >> 16) & 0xFFFFu) | (1u << 16)),
              (int)((elems & 0xFFFFu) << 16),
              0, (int)elems, 0, 0};
  i32x4 z4 = {0, 0, 0, 0};
  i32x8 z8 = {0, 0, 0, 0, 0, 0, 0, 0};
  __builtin_amdgcn_tensor_load_to_lds(g0, g1, z4, z4, z8, 0);
  wait_tensor0();
#else
  const int lane = threadIdx.x & 31;
  const uint4* s = (const uint4*)gsrc;
  uint4* d = (uint4*)lds_dst;
  int n = (int)(elems >> 3);
  for (int i = lane; i < n; i += 32) d[i] = s[i];
#endif
}

static __device__ __forceinline__ v16bf frag16(const bf16_t* p) {
  union { v16bf v; uint4 q[2]; } u;
  u.q[0] = *(const uint4*)(p);
  u.q[1] = *(const uint4*)(p + 8);
  return u.v;
}

// A fragment: lanes 0-15 -> M=lane, K=k0..k0+15 ; lanes 16-31 -> K=k0+16..k0+31
static __device__ __forceinline__ v16bf a_frag(const bf16_t* tile, int ld, int k0) {
  const int lane = threadIdx.x & 31;
  return frag16(tile + (lane & 15) * ld + k0 + ((lane >> 4) << 4));
}

// B fragment from packed weights: block = nt*KSTEPS + ks
static __device__ __forceinline__ v16bf b_frag(const bf16_t* pk, int block) {
  const int lane = threadIdx.x & 31;
  return frag16(pk + ((size_t)((block << 5) + lane) << 4));
}

// ---------------------------------------------------------------------------
// Weight packing: f32 row-major W[K,N] -> bf16 WMMA-B fragment layout.
// ---------------------------------------------------------------------------
__global__ void pack_w_k(const float* __restrict__ W, bf16_t* __restrict__ pk,
                         int K, int N, int Npad) {
  int idx = blockIdx.x * blockDim.x + threadIdx.x;
  int total = K * Npad;
  if (idx >= total) return;
  int j    = idx & 15;
  int lane = (idx >> 4) & 31;
  int blk  = idx >> 9;
  int ksteps = K >> 5;
  int nt = blk / ksteps;
  int ks = blk - nt * ksteps;
  int n = nt * 16 + (lane & 15);
  int k = ks * 32 + ((lane >> 4) << 4) + j;
  float v = (n < N) ? W[(size_t)k * N + n] : 0.0f;
  pk[idx] = (bf16_t)v;
}

__global__ void count_k(const int* __restrict__ dst, float* __restrict__ cnt, int E) {
  int i = blockIdx.x * blockDim.x + threadIdx.x;
  if (i < E) atomicAdd(&cnt[dst[i]], 1.0f);
}

// ---------------------------------------------------------------------------
// h = x @ Wn + bn   : [N,64] @ [64,128], h stored bf16
// ---------------------------------------------------------------------------
__global__ __launch_bounds__(256) void node_proj_k(
    const float* __restrict__ x, const bf16_t* __restrict__ Wpk,
    const float* __restrict__ bias, bf16_t* __restrict__ h, int N) {
  extern __shared__ __align__(16) char smem[];
  const int WAVES = blockDim.x >> 5;
  bf16_t* Wl = (bf16_t*)smem;                  // 64*128
  bf16_t* Ab = Wl + 64 * HID;                  // WAVES*16*64
  bf16_t* Ob = Ab + (size_t)WAVES * 16 * 64;   // WAVES*16*128
  const int tid = threadIdx.x, lane = tid & 31, wave = tid >> 5;

  if (wave == 0) tdm_load_weights(Wpk, Wl, 64 * HID);
  __syncthreads();

  bf16_t* A = Ab + wave * 16 * 64;
  bf16_t* O = Ob + wave * 16 * HID;
  int row0 = (blockIdx.x * WAVES + wave) * 16;
  if (row0 >= N) return;

  {  // stage A: 16 rows x 64 f32 -> bf16
    int r = lane >> 1, half = lane & 1;
    int row = row0 + r; int rc = row < N ? row : N - 1;
    const float4* xp = (const float4*)(x + (size_t)rc * 64 + half * 32);
    bf16_t* ar = A + r * 64 + half * 32;
#pragma unroll
    for (int i = 0; i < 8; ++i) {
      float4 v = xp[i];
      ar[i * 4 + 0] = (bf16_t)v.x; ar[i * 4 + 1] = (bf16_t)v.y;
      ar[i * 4 + 2] = (bf16_t)v.z; ar[i * 4 + 3] = (bf16_t)v.w;
    }
  }

  const int col16 = lane & 15, ro = (lane >> 4) << 3;
  v8f acc[8];
#pragma unroll
  for (int nt = 0; nt < 8; ++nt) {
    float bc = bias[nt * 16 + col16];
    acc[nt] = (v8f){bc, bc, bc, bc, bc, bc, bc, bc};
  }
#pragma unroll
  for (int ks = 0; ks < 2; ++ks) {
    v16bf a = a_frag(A, 64, ks * 32);
#pragma unroll
    for (int nt = 0; nt < 8; ++nt)
      acc[nt] = wmma_bf16(a, b_frag(Wl, nt * 2 + ks), acc[nt]);
  }
  // epilogue via LDS -> coalesced b128 stores
#pragma unroll
  for (int nt = 0; nt < 8; ++nt)
#pragma unroll
    for (int r = 0; r < 8; ++r)
      O[(ro + r) * HID + nt * 16 + col16] = (bf16_t)acc[nt][r];
  if (row0 + 16 <= N) {
    uint4* dst = (uint4*)(h + (size_t)row0 * HID);
    const uint4* src = (const uint4*)O;
#pragma unroll
    for (int i = lane; i < 16 * HID / 8; i += 32) dst[i] = src[i];
  } else {
    for (int rr = 0; rr < 16 && row0 + rr < N; ++rr)
      for (int c = lane; c < HID; c += 32)
        h[(size_t)(row0 + rr) * HID + c] = O[rr * HID + c];
  }
}

// ---------------------------------------------------------------------------
// Fused edge message + scatter-add:
//   m = relu([h[src] || (ea*We+be)] @ W1 + b1) @ W2 + b2 ; s[dst] += m
// ---------------------------------------------------------------------------
__global__ __launch_bounds__(256) void msg_k(
    const bf16_t* __restrict__ h, const float* __restrict__ edge_attr,
    const int* __restrict__ src, const int* __restrict__ dst,
    const bf16_t* __restrict__ W1pk, const float* __restrict__ b1,
    const bf16_t* __restrict__ W2pk, const float* __restrict__ b2,
    const float* __restrict__ We, const float* __restrict__ be,
    float* __restrict__ s, int E) {
  extern __shared__ __align__(16) char smem[];
  const int WAVES = blockDim.x >> 5;
  bf16_t* W1l = (bf16_t*)smem;                 // 256*128
  bf16_t* W2l = W1l + 256 * HID;               // 128*128
  bf16_t* Ab  = W2l + HID * HID;               // WAVES*16*256
  bf16_t* Mb  = Ab + (size_t)WAVES * 16 * 256; // WAVES*16*128
  int*    Db  = (int*)(Mb + (size_t)WAVES * 16 * HID);

  const int tid = threadIdx.x, lane = tid & 31, wave = tid >> 5;
  if (wave == 0)      tdm_load_weights(W1pk, W1l, 256 * HID);
  else if (wave == 1) tdm_load_weights(W2pk, W2l, HID * HID);
  __syncthreads();

  bf16_t* A = Ab + wave * 16 * 256;
  bf16_t* M = Mb + wave * 16 * HID;
  int* D = Db + wave * 16;
  int ebase = (blockIdx.x * WAVES + wave) * 16;
  if (ebase >= E) return;

  {  // gather h[src] via async LDS loads + synthesize e columns
    int er = lane >> 1, half = lane & 1;
    int e = ebase + er; int ec = e < E ? e : E - 1;
    int sn = src[ec];
    const bf16_t* hrow = h + (size_t)sn * HID + half * 64;
    bf16_t* a0 = A + er * 256 + half * 64;
#pragma unroll
    for (int i = 0; i < 8; ++i) async_b128(a0 + i * 8, hrow + i * 8);
    float ea = edge_attr[ec];
    bf16_t* a1 = A + er * 256 + 128 + half * 64;
    const float* wep = We + half * 64;
    const float* bep = be + half * 64;
#pragma unroll
    for (int i = 0; i < 64; ++i) a1[i] = (bf16_t)(ea * wep[i] + bep[i]);
    if (lane < 16) { int ee = ebase + lane; D[lane] = dst[ee < E ? ee : E - 1]; }
    wait_async0();
  }

  const int col16 = lane & 15, ro = (lane >> 4) << 3;
  v8f acc[8];
  // GEMM1: [16,256] @ [256,128] -> relu -> M (bf16)
#pragma unroll
  for (int nt = 0; nt < 8; ++nt) {
    float bc = b1[nt * 16 + col16];
    acc[nt] = (v8f){bc, bc, bc, bc, bc, bc, bc, bc};
  }
#pragma unroll
  for (int ks = 0; ks < 8; ++ks) {
    v16bf a = a_frag(A, 256, ks * 32);
#pragma unroll
    for (int nt = 0; nt < 8; ++nt)
      acc[nt] = wmma_bf16(a, b_frag(W1l, nt * 8 + ks), acc[nt]);
  }
#pragma unroll
  for (int nt = 0; nt < 8; ++nt)
#pragma unroll
    for (int r = 0; r < 8; ++r)
      M[(ro + r) * HID + nt * 16 + col16] = (bf16_t)fmaxf(acc[nt][r], 0.0f);

  // GEMM2: [16,128] @ [128,128] -> scatter atomics into s[dst]
#pragma unroll
  for (int nt = 0; nt < 8; ++nt) {
    float bc = b2[nt * 16 + col16];
    acc[nt] = (v8f){bc, bc, bc, bc, bc, bc, bc, bc};
  }
#pragma unroll
  for (int ks = 0; ks < 4; ++ks) {
    v16bf a = a_frag(M, HID, ks * 32);
#pragma unroll
    for (int nt = 0; nt < 8; ++nt)
      acc[nt] = wmma_bf16(a, b_frag(W2l, nt * 4 + ks), acc[nt]);
  }
  if (ebase + 16 <= E) {
#pragma unroll
    for (int nt = 0; nt < 8; ++nt) {
      int col = nt * 16 + col16;
#pragma unroll
      for (int r = 0; r < 8; ++r)
        atomicAdd(&s[(size_t)D[ro + r] * HID + col], acc[nt][r]);
    }
  } else {
#pragma unroll
    for (int nt = 0; nt < 8; ++nt) {
      int col = nt * 16 + col16;
#pragma unroll
      for (int r = 0; r < 8; ++r)
        if (ebase + ro + r < E)
          atomicAdd(&s[(size_t)D[ro + r] * HID + col], acc[nt][r]);
    }
  }
}

// ---------------------------------------------------------------------------
// Fused update: h = relu([h || s/max(cnt,1)] @ Wu + bu)  (in-place, row-local)
// ---------------------------------------------------------------------------
__global__ __launch_bounds__(256) void upd_k(
    bf16_t* __restrict__ h, const float* __restrict__ s, const float* __restrict__ cnt,
    const bf16_t* __restrict__ Wpk, const float* __restrict__ bias, int N) {
  extern __shared__ __align__(16) char smem[];
  const int WAVES = blockDim.x >> 5;
  bf16_t* Wl = (bf16_t*)smem;                 // 256*128
  bf16_t* Ab = Wl + 256 * HID;                // WAVES*16*256 (reused as out tile)
  const int tid = threadIdx.x, lane = tid & 31, wave = tid >> 5;

  if (wave == 0) tdm_load_weights(Wpk, Wl, 256 * HID);
  __syncthreads();

  bf16_t* A = Ab + wave * 16 * 256;
  int row0 = (blockIdx.x * WAVES + wave) * 16;
  if (row0 >= N) return;

  {  // stage [h || s/max(cnt,1)]
    int r = lane >> 1, half = lane & 1;
    int row = row0 + r; int rc = row < N ? row : N - 1;
    const bf16_t* hrow = h + (size_t)rc * HID + half * 64;
    bf16_t* a0 = A + r * 256 + half * 64;
#pragma unroll
    for (int i = 0; i < 8; ++i) async_b128(a0 + i * 8, hrow + i * 8);
    float inv = 1.0f / fmaxf(cnt[rc], 1.0f);
    const float4* sp = (const float4*)(s + (size_t)rc * HID + half * 64);
    bf16_t* a1 = A + r * 256 + 128 + half * 64;
#pragma unroll
    for (int i = 0; i < 16; ++i) {
      float4 u = sp[i];
      a1[i * 4 + 0] = (bf16_t)(u.x * inv); a1[i * 4 + 1] = (bf16_t)(u.y * inv);
      a1[i * 4 + 2] = (bf16_t)(u.z * inv); a1[i * 4 + 3] = (bf16_t)(u.w * inv);
    }
    wait_async0();
  }

  const int col16 = lane & 15, ro = (lane >> 4) << 3;
  v8f acc[8];
#pragma unroll
  for (int nt = 0; nt < 8; ++nt) {
    float bc = bias[nt * 16 + col16];
    acc[nt] = (v8f){bc, bc, bc, bc, bc, bc, bc, bc};
  }
#pragma unroll
  for (int ks = 0; ks < 8; ++ks) {
    v16bf a = a_frag(A, 256, ks * 32);
#pragma unroll
    for (int nt = 0; nt < 8; ++nt)
      acc[nt] = wmma_bf16(a, b_frag(Wl, nt * 8 + ks), acc[nt]);
  }
  // epilogue: relu -> reuse A as 16x128 out tile -> coalesced stores
  bf16_t* O = A;
#pragma unroll
  for (int nt = 0; nt < 8; ++nt)
#pragma unroll
    for (int r = 0; r < 8; ++r)
      O[(ro + r) * HID + nt * 16 + col16] = (bf16_t)fmaxf(acc[nt][r], 0.0f);
  if (row0 + 16 <= N) {
    uint4* dst = (uint4*)(h + (size_t)row0 * HID);
    const uint4* src = (const uint4*)O;
#pragma unroll
    for (int i = lane; i < 16 * HID / 8; i += 32) dst[i] = src[i];
  } else {
    for (int rr = 0; rr < 16 && row0 + rr < N; ++rr)
      for (int c = lane; c < HID; c += 32)
        h[(size_t)(row0 + rr) * HID + c] = O[rr * HID + c];
  }
}

// ---------------------------------------------------------------------------
// Classifier: out = relu(h[g*NPG] @ Wc1 + bc1) @ Wc2 + bc2   -> [G,10] f32
// ---------------------------------------------------------------------------
__global__ __launch_bounds__(256) void clf_k(
    const bf16_t* __restrict__ h,
    const bf16_t* __restrict__ pkC1, const float* __restrict__ bc1,
    const bf16_t* __restrict__ pkC2, const float* __restrict__ bc2,
    float* __restrict__ out, int G, int NPG) {
  extern __shared__ __align__(16) char smem[];
  const int WAVES = blockDim.x >> 5;
  bf16_t* Ab = (bf16_t*)smem;                   // WAVES*16*128
  bf16_t* Mb = Ab + (size_t)WAVES * 16 * HID;   // WAVES*16*64
  const int tid = threadIdx.x, lane = tid & 31, wave = tid >> 5;
  bf16_t* A = Ab + wave * 16 * HID;
  bf16_t* M = Mb + wave * 16 * 64;
  int g0 = (blockIdx.x * WAVES + wave) * 16;
  if (g0 >= G) return;

  {  // gather obs rows (h[g*NPG]) via async LDS loads
    int r = lane >> 1, half = lane & 1;
    int g = g0 + r; int gc = g < G ? g : G - 1;
    const bf16_t* hrow = h + (size_t)gc * NPG * HID + half * 64;
    bf16_t* ar = A + r * HID + half * 64;
#pragma unroll
    for (int i = 0; i < 8; ++i) async_b128(ar + i * 8, hrow + i * 8);
    wait_async0();
  }

  const int col16 = lane & 15, ro = (lane >> 4) << 3;
  // GEMM1: [16,128] @ [128,64] -> relu -> M
  v8f acc[4];
#pragma unroll
  for (int nt = 0; nt < 4; ++nt) {
    float bc = bc1[nt * 16 + col16];
    acc[nt] = (v8f){bc, bc, bc, bc, bc, bc, bc, bc};
  }
#pragma unroll
  for (int ks = 0; ks < 4; ++ks) {
    v16bf a = a_frag(A, HID, ks * 32);
#pragma unroll
    for (int nt = 0; nt < 4; ++nt)
      acc[nt] = wmma_bf16(a, b_frag(pkC1, nt * 4 + ks), acc[nt]);
  }
#pragma unroll
  for (int nt = 0; nt < 4; ++nt)
#pragma unroll
    for (int r = 0; r < 8; ++r)
      M[(ro + r) * 64 + nt * 16 + col16] = (bf16_t)fmaxf(acc[nt][r], 0.0f);

  // GEMM2: [16,64] @ [64,16(pad of 10)]
  {
    float bc = (col16 < 10) ? bc2[col16] : 0.0f;
    v8f c = {bc, bc, bc, bc, bc, bc, bc, bc};
#pragma unroll
    for (int ks = 0; ks < 2; ++ks)
      c = wmma_bf16(a_frag(M, 64, ks * 32), b_frag(pkC2, ks), c);
    if (col16 < 10) {
#pragma unroll
      for (int r = 0; r < 8; ++r) {
        int g = g0 + ro + r;
        if (g < G) out[(size_t)g * 10 + col16] = c[r];
      }
    }
  }
}

// ---------------------------------------------------------------------------
// Host orchestration
// ---------------------------------------------------------------------------
extern "C" void kernel_launch(void* const* d_in, const int* in_sizes, int n_in,
                              void* d_out, int out_size, void* d_ws, size_t ws_size,
                              hipStream_t stream) {
  (void)n_in; (void)ws_size;
  const float* x     = (const float*)d_in[0];
  const float* eattr = (const float*)d_in[1];
  const float* Wn    = (const float*)d_in[2];
  const float* bn    = (const float*)d_in[3];
  const float* We    = (const float*)d_in[4];
  const float* be    = (const float*)d_in[5];
  const float *W1[3], *b1[3], *W2[3], *b2[3], *Wu[3], *bu[3];
  for (int i = 0; i < 3; ++i) {
    int base = 6 + i * 6;
    W1[i] = (const float*)d_in[base + 0]; b1[i] = (const float*)d_in[base + 1];
    W2[i] = (const float*)d_in[base + 2]; b2[i] = (const float*)d_in[base + 3];
    Wu[i] = (const float*)d_in[base + 4]; bu[i] = (const float*)d_in[base + 5];
  }
  const float* Wc1 = (const float*)d_in[24];
  const float* bc1 = (const float*)d_in[25];
  const float* Wc2 = (const float*)d_in[26];
  const float* bc2 = (const float*)d_in[27];
  const int*   ei  = (const int*)d_in[28];

  const int N = in_sizes[0] / 64;
  const int E = in_sizes[1];
  const int G = out_size / 10;
  const int NPG = N / G;
  const int* srcI = ei;
  const int* dstI = ei + E;
  float* out = (float*)d_out;

  char* ws = (char*)d_ws;
  size_t off = 0;
  auto carve = [&](size_t bytes) -> void* {
    void* p = ws + off;
    off = (off + bytes + 255) & ~(size_t)255;
    return p;
  };
  bf16_t* h    = (bf16_t*)carve((size_t)N * HID * 2);
  float*  sbuf = (float*)carve((size_t)N * HID * 4);
  float*  cnt  = (float*)carve((size_t)N * 4);
  bf16_t* pkN  = (bf16_t*)carve(64 * HID * 2);
  bf16_t *pkW1[3], *pkW2[3], *pkWu[3];
  for (int i = 0; i < 3; ++i) {
    pkW1[i] = (bf16_t*)carve(256 * HID * 2);
    pkW2[i] = (bf16_t*)carve(HID * HID * 2);
    pkWu[i] = (bf16_t*)carve(256 * HID * 2);
  }
  bf16_t* pkC1 = (bf16_t*)carve(HID * 64 * 2);
  bf16_t* pkC2 = (bf16_t*)carve(64 * 16 * 2);

  auto pack = [&](const float* W, bf16_t* pk, int K, int Nw, int Npad) {
    int total = K * Npad;
    pack_w_k<<<(total + 255) / 256, 256, 0, stream>>>(W, pk, K, Nw, Npad);
  };
  pack(Wn, pkN, 64, HID, HID);
  for (int i = 0; i < 3; ++i) {
    pack(W1[i], pkW1[i], 256, HID, HID);
    pack(W2[i], pkW2[i], HID, HID, HID);
    pack(Wu[i], pkWu[i], 256, HID, HID);
  }
  pack(Wc1, pkC1, HID, 64, 64);
  pack(Wc2, pkC2, 64, 10, 16);

  (void)hipMemsetAsync(cnt, 0, (size_t)N * 4, stream);
  count_k<<<(E + 255) / 256, 256, 0, stream>>>(dstI, cnt, E);

  const int WAVES = 8, BLK = 256;
  {  // node projection
    int tiles = (N + 15) / 16;
    int blocks = (tiles + WAVES - 1) / WAVES;
    size_t smem = (size_t)64 * HID * 2 + (size_t)WAVES * 16 * 64 * 2 +
                  (size_t)WAVES * 16 * HID * 2;
    (void)hipFuncSetAttribute((const void*)node_proj_k,
                              hipFuncAttributeMaxDynamicSharedMemorySize, (int)smem);
    node_proj_k<<<blocks, BLK, smem, stream>>>(x, pkN, bn, h, N);
  }

  size_t smemMsg = (size_t)256 * HID * 2 + (size_t)HID * HID * 2 +
                   (size_t)WAVES * 16 * 256 * 2 + (size_t)WAVES * 16 * HID * 2 +
                   (size_t)WAVES * 16 * 4;
  size_t smemUpd = (size_t)256 * HID * 2 + (size_t)WAVES * 16 * 256 * 2;
  (void)hipFuncSetAttribute((const void*)msg_k,
                            hipFuncAttributeMaxDynamicSharedMemorySize, (int)smemMsg);
  (void)hipFuncSetAttribute((const void*)upd_k,
                            hipFuncAttributeMaxDynamicSharedMemorySize, (int)smemUpd);

  for (int l = 0; l < 3; ++l) {
    (void)hipMemsetAsync(sbuf, 0, (size_t)N * HID * 4, stream);
    {
      int tiles = (E + 15) / 16;
      int blocks = (tiles + WAVES - 1) / WAVES;
      msg_k<<<blocks, BLK, smemMsg, stream>>>(h, eattr, srcI, dstI,
                                              pkW1[l], b1[l], pkW2[l], b2[l],
                                              We, be, sbuf, E);
    }
    {
      int tiles = (N + 15) / 16;
      int blocks = (tiles + WAVES - 1) / WAVES;
      upd_k<<<blocks, BLK, smemUpd, stream>>>(h, sbuf, cnt, pkWu[l], bu[l], N);
    }
  }

  {  // classifier
    int tiles = (G + 15) / 16;
    int blocks = (tiles + WAVES - 1) / WAVES;
    size_t smem = (size_t)WAVES * 16 * HID * 2 + (size_t)WAVES * 16 * 64 * 2;
    (void)hipFuncSetAttribute((const void*)clf_k,
                              hipFuncAttributeMaxDynamicSharedMemorySize, (int)smem);
    clf_k<<<blocks, BLK, smem, stream>>>(h, pkC1, bc1, pkC2, bc2, out, G, NPG);
  }
}
